// CTCDecoder_23132693856405
// MI455X (gfx1250) — matrix operand order repaired and checked by reading, not compiled, not run on hardware
//
#include <hip/hip_runtime.h>
#include <hip/hip_bf16.h>

typedef __bf16 bf16;
typedef __attribute__((ext_vector_type(8)))  __bf16 v8bf;
typedef __attribute__((ext_vector_type(16))) __bf16 v16bf;
typedef __attribute__((ext_vector_type(8)))  float  v8f;

#define T_FRAMES 800
#define BATCH    16
#define DIM      256
#define VOC      4001
#define NPAD     4032               // 63*64, multiple of 16; padded cols are zero
#define M_ROWS   (T_FRAMES*BATCH)   // 12800
#define ULAB     100
#define SEXT     201                // 2*U+1
#define NEGINF   (-1e30f)

// ---------------------------------------------------------------- conversions
__global__ void conv_ctx_bf16(const float* __restrict__ x, bf16* __restrict__ o, int n) {
  int i = blockIdx.x * blockDim.x + threadIdx.x;
  if (i < n) o[i] = (bf16)x[i];
}

// Wt[n][k] = W[k][n] (bf16), rows n in [VOC, NPAD) zero-filled.
__global__ void conv_w_bf16(const float* __restrict__ W, bf16* __restrict__ Wt) {
  int i = blockIdx.x * blockDim.x + threadIdx.x;   // over NPAD*DIM
  int n = i / DIM, k = i % DIM;
  Wt[(size_t)n * DIM + k] = (n < VOC) ? (bf16)W[(size_t)k * VOC + n] : (bf16)0.0f;
}

// ---------------------------------------------------------------- WMMA GEMM
// out[m, n] = sum_k A[m,k]*Wt[n,k] + bias[n]; A:[M,256] bf16, Wt:[NPAD,256] bf16
// Block = 256 threads = 8 waves; block tile 128(M) x 16(N); wave tile 16x16.
__global__ __launch_bounds__(256) void gemm_logits(const bf16* __restrict__ A,
                                                   const bf16* __restrict__ Bt,
                                                   const float* __restrict__ bias,
                                                   float* __restrict__ out) {
  const int tid  = threadIdx.x;
  const int wave = tid >> 5;
  const int lane = tid & 31;
  const int hlf  = lane >> 4;     // lane half selects K sub-block / M sub-rows
  const int lid  = lane & 15;

  const int m0 = blockIdx.x * 128 + wave * 16;
  const int n0 = blockIdx.y * 16;

  const bf16* arow = A  + (size_t)(m0 + lid) * DIM;  // A row for this lane
  const bf16* brow = Bt + (size_t)(n0 + lid) * DIM;  // B column (row of Wt)
  const int offa = hlf * 8;    // A 16-bit layout: K chunks {off..off+7, off+16..off+23}
  const int offb = hlf * 8;    // B 16-bit layout: K chunks {off..off+7, off+16..off+23}

  v8f acc = {0.f, 0.f, 0.f, 0.f, 0.f, 0.f, 0.f, 0.f};

#pragma unroll
  for (int kt = 0; kt < DIM; kt += 32) {
    v8bf a_lo = *(const v8bf*)(arow + kt + offa);
    v8bf a_hi = *(const v8bf*)(arow + kt + offa + 16);
    v8bf b_lo = *(const v8bf*)(brow + kt + offb);
    v8bf b_hi = *(const v8bf*)(brow + kt + offb + 16);
    v16bf af = __builtin_shufflevector(a_lo, a_hi, 0,1,2,3,4,5,6,7,8,9,10,11,12,13,14,15);
    v16bf bb = __builtin_shufflevector(b_lo, b_hi, 0,1,2,3,4,5,6,7,8,9,10,11,12,13,14,15);
    acc = __builtin_amdgcn_wmma_f32_16x16x32_bf16(false, af, false, bb,
                                                  (short)0, acc, false, false);
  }

  const int n = n0 + lid;          // all 8 acc values in a lane share column n
  if (n < VOC) {
    const float bn = bias[n];
#pragma unroll
    for (int r = 0; r < 8; ++r) {
      const int m = m0 + r + 8 * hlf;
      out[(size_t)m * VOC + n] = acc[r] + bn;
    }
  }
}

// ---------------------------------------------------------- in-place log-softmax
// One block per row; 16 elements per thread held in registers (single read+write).
__global__ __launch_bounds__(256) void logsoftmax_rows(float* __restrict__ out) {
  __shared__ float red[256];
  const int m = blockIdx.x, tid = threadIdx.x;
  float* row = out + (size_t)m * VOC;

  float v[16];
  float mx = NEGINF;
#pragma unroll
  for (int j = 0; j < 16; ++j) {
    int i = tid + j * 256;
    v[j] = (i < VOC) ? row[i] : NEGINF;
    mx = fmaxf(mx, v[j]);
  }
  red[tid] = mx; __syncthreads();
  for (int s = 128; s > 0; s >>= 1) {
    if (tid < s) red[tid] = fmaxf(red[tid], red[tid + s]);
    __syncthreads();
  }
  mx = red[0]; __syncthreads();

  float sum = 0.f;
#pragma unroll
  for (int j = 0; j < 16; ++j) {
    int i = tid + j * 256;
    if (i < VOC) sum += __expf(v[j] - mx);
  }
  red[tid] = sum; __syncthreads();
  for (int s = 128; s > 0; s >>= 1) {
    if (tid < s) red[tid] += red[tid + s];
    __syncthreads();
  }
  const float lse = mx + __logf(red[0]);

#pragma unroll
  for (int j = 0; j < 16; ++j) {
    int i = tid + j * 256;
    if (i < VOC) row[i] = v[j] - lse;
  }
}

// ---------------------------------------------------------------- CTC recursion
__device__ __forceinline__ float lae(float a, float b) {
  float m = fmaxf(a, b);
  float d = fminf(a, b) - m;
  return m + __logf(1.0f + __expf(d));
}

// One block per batch element; alpha[201] double-buffered in LDS; serial in T.
__global__ __launch_bounds__(256) void ctc_alpha(const float* __restrict__ lp,
                                                 const int* __restrict__ y,
                                                 float* __restrict__ ll_out) {
  __shared__ float bufA[SEXT + 8], bufB[SEXT + 8];
  __shared__ int exts[SEXT];
  __shared__ unsigned char allow[SEXT];
  __shared__ int len_s;

  const int b = blockIdx.x, tid = threadIdx.x;

  if (tid == 0) {
    int len = 0;
    for (int u = 0; u < ULAB; ++u) len += (y[u * BATCH + b] != 1) ? 1 : 0;
    len_s = len;
  }
  for (int s = tid; s < SEXT; s += 256)
    exts[s] = (s & 1) ? y[((s - 1) >> 1) * BATCH + b] : 0;
  __syncthreads();
  for (int s = tid; s < SEXT; s += 256)
    allow[s] = (s >= 2) && (exts[s] != 0) && (exts[s] != exts[s - 2]);
  if (tid < SEXT)
    bufA[tid] = (tid < 2) ? lp[(size_t)b * VOC + exts[tid]] : NEGINF;
  __syncthreads();

  float* ap = bufA;
  float* an = bufB;
  for (int t = 1; t < T_FRAMES; ++t) {
    const size_t base = (size_t)t * (BATCH * VOC) + (size_t)b * VOC;
    if (tid < SEXT) {
      const float e  = lp[base + exts[tid]];          // issue gather early
      const float a0 = ap[tid];
      const float a1 = (tid >= 1) ? ap[tid - 1] : NEGINF;
      const float a2 = allow[tid] ? ap[tid - 2] : NEGINF;
      an[tid] = lae(lae(a0, a1), a2) + e;
    }
    __syncthreads();
    float* tsw = ap; ap = an; an = tsw;
  }

  if (tid == 0) {
    const int e = 2 * len_s;
    ll_out[b] = lae(ap[e], ap[e - 1]);
  }
}

__global__ void finalize_loss(const float* __restrict__ ll, float* __restrict__ loss) {
  if (threadIdx.x == 0) {
    float s = 0.f;
    for (int i = 0; i < BATCH; ++i) s += ll[i];
    loss[0] = -s;
  }
}

// ---------------------------------------------------------------- launcher
extern "C" void kernel_launch(void* const* d_in, const int* in_sizes, int n_in,
                              void* d_out, int out_size, void* d_ws, size_t ws_size,
                              hipStream_t stream) {
  const float* ctx  = (const float*)d_in[0];   // [800,16,256] f32
  const int*   y    = (const int*)d_in[1];     // [100,16] i32
  const float* W    = (const float*)d_in[2];   // [256,4001] f32
  const float* bias = (const float*)d_in[3];   // [4001] f32

  float* out   = (float*)d_out;
  float* logps = out + 1;                      // d_out = [loss | logps(T,B,V)]

  // workspace layout (256B aligned slices)
  char* ws = (char*)d_ws;
  const size_t szA = (size_t)M_ROWS * DIM * sizeof(bf16);        // 6.55 MB
  const size_t szW = (size_t)NPAD  * DIM * sizeof(bf16);         // 2.06 MB
  bf16*  Abf = (bf16*)ws;
  bf16*  Wt  = (bf16*)(ws + ((szA + 255) & ~(size_t)255));
  float* ll  = (float*)(ws + ((szA + 255) & ~(size_t)255) + ((szW + 255) & ~(size_t)255));

  conv_ctx_bf16<<<(M_ROWS * DIM) / 256, 256, 0, stream>>>(ctx, Abf, M_ROWS * DIM);
  conv_w_bf16<<<(NPAD * DIM) / 256, 256, 0, stream>>>(W, Wt);

  dim3 ggrid(M_ROWS / 128, NPAD / 16);         // 100 x 252 tiles
  gemm_logits<<<ggrid, 256, 0, stream>>>(Abf, Wt, bias, logps);

  logsoftmax_rows<<<M_ROWS, 256, 0, stream>>>(logps);

  ctc_alpha<<<BATCH, 256, 0, stream>>>(logps, y, ll);
  finalize_loss<<<1, 32, 0, stream>>>(ll, out);
}
